// Seq2SeqASR_60541859004865
// MI455X (gfx1250) — matrix operand order repaired and checked by reading, not compiled, not run on hardware
//
#include <hip/hip_runtime.h>

// ---------- types ----------
typedef __attribute__((ext_vector_type(16))) _Float16 h16;
typedef __attribute__((ext_vector_type(8)))  _Float16 h8;
typedef __attribute__((ext_vector_type(8)))  float    f8;
typedef __attribute__((ext_vector_type(4)))  unsigned int u32x4;
typedef __attribute__((ext_vector_type(8)))  int i32x8;
typedef __attribute__((ext_vector_type(4)))  int i32x4;

// ---------- optional Tensor Data Mover (CDNA5 TDM) ----------
#if defined(__has_builtin)
#if __has_builtin(__builtin_amdgcn_tensor_load_to_lds) && __has_builtin(__builtin_amdgcn_s_wait_tensorcnt)
#define USE_TDM 1
#endif
#endif

#if defined(USE_TDM)
// 1-D contiguous copy: nelem f16 elements from gsrc -> LDS byte offset lds_off.
// D# packing per CDNA5 ISA ch.8 (group0: count/lds_addr/global_addr/type,
// group1: data_size=2B, tensor_dim0/1, tile_dim0/1, stride0).
__device__ __forceinline__ void tdm_load_1d_f16(const void* gsrc, unsigned lds_off,
                                                unsigned nelem) {
    unsigned long long ga = (unsigned long long)gsrc;
    u32x4 g0;
    g0[0] = 1u;                                            // count=1, user descriptor
    g0[1] = lds_off;                                       // lds_addr (bytes)
    g0[2] = (unsigned)(ga & 0xFFFFFFFFu);                  // global_addr[31:0]
    g0[3] = (unsigned)((ga >> 32) & 0x01FFFFFFu)           // global_addr[56:32]
          | (2u << 30);                                    // type=2 ("image")
    i32x8 g1;
    g1[0] = 0x00010000;                                    // data_size=1 -> 2 bytes
    g1[1] = (int)((nelem & 0xFFFFu) << 16);                // tensor_dim0[15:0]
    g1[2] = (int)(((nelem >> 16) & 0xFFFFu) | (1u << 16)); // tensor_dim0[31:16], tensor_dim1=1
    g1[3] = (int)((nelem & 0xFFFFu) << 16);                // tile_dim0 = nelem (<=65535)
    g1[4] = 1;                                             // tile_dim1 = 1
    g1[5] = (int)nelem;                                    // tensor_dim0_stride low
    g1[6] = 0;
    g1[7] = 0;
    i32x4 z = {0, 0, 0, 0};
#if __clang_major__ >= 23
    i32x8 z8 = {0, 0, 0, 0, 0, 0, 0, 0};
    __builtin_amdgcn_tensor_load_to_lds(g0, g1, z, z, z8, 0);
#else
    __builtin_amdgcn_tensor_load_to_lds(g0, g1, z, z, 0);
#endif
}
#endif

// ---------- WMMA helpers (gfx1250, wave32) ----------
// A fragment: 16x32 f16, row m = lane&15, group g = lane>>4
//   element e<8 :  k = kk + g*8 + e
//   element e>=8:  k = kk + 16 + g*8 + (e-8)
__device__ __forceinline__ h16 load_a_frag(const _Float16* A, int lda, int kk) {
    int lane = threadIdx.x & 31;
    int m = lane & 15;
    int g = lane >> 4;
    const _Float16* p = A + (size_t)m * lda + kk + g * 8;
    h8 lo = *(const h8*)p;
    h8 hi = *(const h8*)(p + 16);
    h16 f;
#pragma unroll
    for (int i = 0; i < 8; ++i) { f[i] = lo[i]; f[i + 8] = hi[i]; }
    return f;
}

// B fragment: 32x16 f16 from weight stored (N,K) row-major.
//   col n = lane&15, e<8: k = kk + (lane>>4)*8 + e; e>=8: k = kk + 16 + (lane>>4)*8 + (e-8)
__device__ __forceinline__ h16 load_b_frag(const _Float16* W, int ldw, int kk) {
    int lane = threadIdx.x & 31;
    int n = lane & 15;
    int g = lane >> 4;
    const _Float16* p = W + (size_t)n * ldw + kk + g * 8;
    h8 lo = *(const h8*)p;
    h8 hi = *(const h8*)(p + 16);
    h16 f;
#pragma unroll
    for (int i = 0; i < 8; ++i) { f[i] = lo[i]; f[i + 8] = hi[i]; }
    return f;
}

__device__ __forceinline__ f8 wmma_f16f32(h16 a, h16 b, f8 c) {
    return __builtin_amdgcn_wmma_f32_16x16x32_f16(false, a, false, b, (short)0, c, false, false);
}

// ---------- constants ----------
#define NB   16     // batch
#define HD   512    // hidden
#define TP   200    // encoder time steps
#define EMB  256    // embedding dim
#define VOC  29     // vocab
#define LSEQ 100    // target length

// ---------- f32 -> f16 strided convert ----------
__global__ void cvt_kernel(const float* __restrict__ src, _Float16* __restrict__ dst,
                           int rows, int cols, int stride, int off) {
    int i = blockIdx.x * blockDim.x + threadIdx.x;
    if (i >= rows * cols) return;
    int r = i / cols, c = i % cols;
    dst[i] = (_Float16)src[(size_t)r * stride + off + c];
}

// ---------- conv1 (Cin=1) + BN + ReLU + 2x2 maxpool, fp32 direct ----------
// in (16,1,128,1600) f32 -> out (16,32,64,800) f16
__global__ void conv1_kernel(const float* __restrict__ in, const float* __restrict__ w,
                             const float* __restrict__ cb, const float* __restrict__ gg,
                             const float* __restrict__ be, const float* __restrict__ mm,
                             const float* __restrict__ vv, _Float16* __restrict__ out) {
    int idx = blockIdx.x * blockDim.x + threadIdx.x;
    const int total = 16 * 32 * 64 * 800;
    if (idx >= total) return;
    int px = idx % 800; int t = idx / 800;
    int py = t % 64; t /= 64;
    int c = t % 32; int b = t / 32;
    const float* ib = in + (size_t)b * 128 * 1600;
    const float* wc = w + c * 9;
    float scale = gg[c] * rsqrtf(vv[c] + 1e-5f);
    float mx = -1e30f;
#pragma unroll
    for (int dy = 0; dy < 2; ++dy)
#pragma unroll
        for (int dx = 0; dx < 2; ++dx) {
            int y = py * 2 + dy, x = px * 2 + dx;
            float acc = cb[c];
#pragma unroll
            for (int r = 0; r < 3; ++r) {
                int yy = y + r - 1; if (yy < 0 || yy >= 128) continue;
#pragma unroll
                for (int s = 0; s < 3; ++s) {
                    int xx = x + s - 1; if (xx < 0 || xx >= 1600) continue;
                    acc += ib[yy * 1600 + xx] * wc[r * 3 + s];
                }
            }
            float v = (acc - mm[c]) * scale + be[c];
            v = v > 0.f ? v : 0.f;
            mx = fmaxf(mx, v);
        }
    out[idx] = (_Float16)mx;
}

// ---------- implicit-GEMM conv (3x3 SAME) via WMMA + fused BN/ReLU ----------
// All shape parameters are compile-time: div/mod strength-reduce, K-loop unrolls.
template <int Cin, int H, int W>
__global__ void conv_gemm_kernel(const _Float16* __restrict__ act, const _Float16* __restrict__ wf,
                                 const float* __restrict__ cb, const float* __restrict__ gg,
                                 const float* __restrict__ be, const float* __restrict__ mm,
                                 const float* __restrict__ vv, _Float16* __restrict__ out,
                                 int Cout) {
    constexpr int K = Cin * 9;
    constexpr int PT = (H * W) >> 4;
    extern __shared__ char smc[];
    _Float16* As = (_Float16*)smc;
    int b  = blockIdx.x / PT;
    int pt = blockIdx.x % PT;
    int nt = threadIdx.x >> 5;
    int lane = threadIdx.x & 31;
    const _Float16* ab = act + (size_t)b * Cin * H * W;

    // cooperative im2col staging: one 16xK tile per block (compile-time div/mod)
    for (int e = threadIdx.x; e < 16 * K; e += blockDim.x) {
        int mrow = e / K;
        int k = e - mrow * K;
        int ci = k / 9; int rs = k - ci * 9; int r = rs / 3; int s = rs - r * 3;
        int pp = pt * 16 + mrow;
        int yy = pp / W + r - 1, xx = pp % W + s - 1;
        _Float16 v = (_Float16)0.f;
        if (yy >= 0 && yy < H && xx >= 0 && xx < W)
            v = ab[((size_t)ci * H + yy) * W + xx];
        As[e] = v;
    }
    __syncthreads();

    f8 acc = {};
    const _Float16* Wn = wf + (size_t)nt * 16 * K;
#pragma unroll
    for (int kk = 0; kk < K; kk += 32) {
        h16 a = load_a_frag(As, K, kk);     // ds_load from staged tile
        h16 bf = load_b_frag(Wn, K, kk);
        acc = wmma_f16f32(a, bf, acc);
    }
    int coln = lane & 15, gid = lane >> 4;
    int c = nt * 16 + coln;
    float scale = gg[c] * rsqrtf(vv[c] + 1e-5f);
    float bias = cb[c], mu = mm[c], bt = be[c];
#pragma unroll
    for (int r = 0; r < 8; ++r) {
        int row = r + gid * 8;
        int pp = pt * 16 + row;
        int yy = pp / W, xx = pp % W;
        float v = ((acc[r] + bias) - mu) * scale + bt;
        v = v > 0.f ? v : 0.f;
        out[(((size_t)b * Cout + c) * H + yy) * W + xx] = (_Float16)v;
    }
}

// ---------- 2x2 maxpool, f16 (compile-time shapes) ----------
template <int C, int H, int W>
__global__ void pool_kernel(const _Float16* __restrict__ in, _Float16* __restrict__ out) {
    constexpr int Wo = W >> 1, Ho = H >> 1;
    constexpr int total = 16 * C * Ho * Wo;
    int idx = blockIdx.x * blockDim.x + threadIdx.x;
    if (idx >= total) return;
    int x = idx % Wo; int t = idx / Wo;
    int y = t % Ho; t /= Ho;
    int c = t % C; int b = t / C;
    const _Float16* p = in + (((size_t)(b * C + c) * H) + y * 2) * W + x * 2;
    float m = fmaxf(fmaxf((float)p[0], (float)p[1]), fmaxf((float)p[W], (float)p[W + 1]));
    out[idx] = (_Float16)m;
}

// ---------- conv3 pool + transpose to (B, T=200, 2048): d = c*16 + f ----------
__global__ void pool3t_kernel(const _Float16* __restrict__ in /*16,128,32,400*/,
                              _Float16* __restrict__ x /*16,200,2048*/) {
    int idx = blockIdx.x * blockDim.x + threadIdx.x;
    const int total = 16 * 200 * 2048;
    if (idx >= total) return;
    int d = idx % 2048; int t = (idx / 2048) % 200; int b = idx / (2048 * 200);
    int c = d >> 4; int f = d & 15;
    const _Float16* p = in + (((size_t)(b * 128 + c) * 32) + f * 2) * 400 + t * 2;
    float m = fmaxf(fmaxf((float)p[0], (float)p[1]), fmaxf((float)p[400], (float)p[401]));
    x[idx] = (_Float16)m;
}

// ---------- generic WMMA GEMM: C(M,N) = A(M,K) * Wt(N,K)^T + bias ----------
// grid.x = M/16, blockDim = 256 (8 waves); A-tile staged in LDS via TDM.
// dynamic LDS: 16*K*2 bytes; N,K compile-time.
template <int N, int K>
__global__ void gemm16_kernel(const _Float16* __restrict__ A, const _Float16* __restrict__ Wt,
                              const float* __restrict__ bias, float* __restrict__ Cf,
                              _Float16* __restrict__ Ch) {
    extern __shared__ char sm16[];
    _Float16* As = (_Float16*)sm16;
    int mt = blockIdx.x;
    int wv = threadIdx.x >> 5;
    int lane = threadIdx.x & 31;
    constexpr int NT = N >> 4;
    const _Float16* Am = A + (size_t)mt * 16 * K;

    // stage contiguous 16xK A-tile into LDS
#if defined(USE_TDM)
    if (wv == 0) {
        tdm_load_1d_f16(Am, 0u, (unsigned)(16 * K));      // tensor_load_to_lds (TENSORcnt)
        __builtin_amdgcn_s_wait_tensorcnt(0);
    }
    __syncthreads();
#else
    {
        constexpr int cnt = (16 * K) >> 3;
        const h8* src = (const h8*)Am;
        h8* dst = (h8*)As;
        for (int i = threadIdx.x; i < cnt; i += blockDim.x) dst[i] = src[i];
        __syncthreads();
    }
#endif

    for (int nt = wv; nt < NT; nt += 8) {
        f8 acc = {};
        const _Float16* Wn = Wt + (size_t)nt * 16 * K;
#pragma unroll 4
        for (int kk = 0; kk < K; kk += 32) {
            if (kk + 64 < K) __builtin_prefetch(Wn + kk + 64, 0, 1);
            h16 a = load_a_frag(As, K, kk);   // ds_load
            h16 b = load_b_frag(Wn, K, kk);   // global_load
            acc = wmma_f16f32(a, b, acc);
        }
        int coln = lane & 15, gid = lane >> 4;
        int n = nt * 16 + coln;
        float bs = bias ? bias[n] : 0.f;
#pragma unroll
        for (int r = 0; r < 8; ++r) {
            int row = mt * 16 + r + gid * 8;
            float v = acc[r] + bs;
            if (Cf) Cf[(size_t)row * N + n] = v;
            if (Ch) Ch[(size_t)row * N + n] = (_Float16)v;
        }
    }
}

// ---------- enc mean over t ----------
__global__ void encmean_kernel(const _Float16* __restrict__ enc, float* __restrict__ em) {
    int idx = blockIdx.x * blockDim.x + threadIdx.x;
    if (idx >= NB * HD) return;
    int b = idx / HD, d = idx % HD;
    float s = 0.f;
    for (int t = 0; t < TP; ++t) s += (float)enc[((size_t)(b * TP + t)) * HD + d];
    em[idx] = s * (1.f / (float)TP);
}

// ---------- h0 = tanh(mean @ init_w^T + init_b) ----------
__global__ void h0_kernel(const float* __restrict__ em, const float* __restrict__ iw,
                          const float* __restrict__ ib, float* __restrict__ h0) {
    int idx = blockIdx.x * blockDim.x + threadIdx.x;
    if (idx >= NB * HD) return;
    int b = idx / HD, i = idx % HD;
    float s = ib[i];
    const float* e = em + (size_t)b * HD;
    const float* w = iw + (size_t)i * HD;
    for (int k = 0; k < HD; ++k) s += e[k] * w[k];
    h0[idx] = tanhf(s);
}

// ---------- persistent attention-GRU decoder: grid=1, 1024 threads (32 waves) ----------
__global__ void __launch_bounds__(1024, 1)
decoder_kernel(const _Float16* __restrict__ enc, const float* __restrict__ enc_proj,
               const float* __restrict__ h0, const _Float16* __restrict__ WhT,
               const float* __restrict__ attn_b, const float* __restrict__ v_w,
               const _Float16* __restrict__ embh, const int* __restrict__ ttext,
               const _Float16* __restrict__ wih, const _Float16* __restrict__ whh,
               const float* __restrict__ bih, const float* __restrict__ bhh,
               const float* __restrict__ out_w, const float* __restrict__ out_b,
               float* __restrict__ preds) {
    extern __shared__ char smem[];
    float*    h    = (float*)smem;                 // 16*512 f32
    _Float16* hf   = (_Float16*)(h + NB * HD);     // 16*512 f16
    float*    attw = (float*)(hf + NB * HD);       // 16*200 f32
    _Float16* xin  = (_Float16*)(attw + NB * TP);  // 16*768 f16
    float*    buf  = (float*)(xin + NB * 768);     // 16*2048 f32 (sWh | S/gin/ghn)

    int tid = threadIdx.x;
    int wv = tid >> 5, lane = tid & 31;

    for (int i = tid; i < NB * HD; i += 1024) {
        float v = h0[i]; h[i] = v; hf[i] = (_Float16)v;
    }
    __syncthreads();

    for (int s = 0; s < LSEQ - 1; ++s) {
        // --- A: sWh[b][j] = h @ Wh^T + attn_b  (WMMA, 32 waves x 1 N-tile) ---
        {
            int nt = wv;  // N=512 -> 32 tiles
            f8 acc = {};
            const _Float16* Wn = WhT + (size_t)nt * 16 * HD;
#pragma unroll 4
            for (int kk = 0; kk < HD; kk += 32) {
                h16 a = load_a_frag(hf, HD, kk);
                h16 b = load_b_frag(Wn, HD, kk);
                acc = wmma_f16f32(a, b, acc);
            }
            int coln = lane & 15, gid = lane >> 4;
            int j = nt * 16 + coln;
            float ab = attn_b[j];
#pragma unroll
            for (int r = 0; r < 8; ++r) buf[(r + gid * 8) * HD + j] = acc[r] + ab;
        }
        __syncthreads();

        // --- B: scores[b][t] = v_w . tanh(enc_proj[b,t,:] + sWh[b,:]) ---
        for (int idx = tid; idx < NB * TP; idx += 1024) {
            int bb = idx / TP, t = idx % TP;
            const float* ep = enc_proj + ((size_t)(bb * TP + t)) * HD;
            const float* sw = buf + (size_t)bb * HD;
            float sc = 0.f;
            for (int k = 0; k < HD; ++k) sc += v_w[k] * tanhf(ep[k] + sw[k]);
            attw[idx] = sc;
        }
        __syncthreads();

        // --- C: softmax over t (one wave per batch row) ---
        if (wv < NB) {
            int bb = wv;
            float mx = -1e30f;
            for (int t = lane; t < TP; t += 32) mx = fmaxf(mx, attw[bb * TP + t]);
            for (int o = 16; o > 0; o >>= 1) mx = fmaxf(mx, __shfl_xor(mx, o, 32));
            float sum = 0.f;
            for (int t = lane; t < TP; t += 32) {
                float e = expf(attw[bb * TP + t] - mx);
                attw[bb * TP + t] = e; sum += e;
            }
            for (int o = 16; o > 0; o >>= 1) sum += __shfl_xor(sum, o, 32);
            float inv = 1.f / sum;
            for (int t = lane; t < TP; t += 32) attw[bb * TP + t] *= inv;
        }
        __syncthreads();

        // --- D: context + embedding -> xin (16 x 768 f16) ---
        for (int idx = tid; idx < NB * HD; idx += 1024) {
            int bb = idx / HD, d = idx % HD;
            const _Float16* eb = enc + ((size_t)bb * TP) * HD + d;
            const float* w = attw + bb * TP;
            float c = 0.f;
            for (int t = 0; t < TP; ++t) c += w[t] * (float)eb[(size_t)t * HD];
            xin[bb * 768 + EMB + d] = (_Float16)c;
        }
        for (int idx = tid; idx < NB * EMB; idx += 1024) {
            int bb = idx / EMB, e = idx % EMB;
            int tok = ttext[bb * LSEQ + s];
            xin[bb * 768 + e] = embh[tok * EMB + e];
        }
        __syncthreads();

        // --- E1: gi = xin @ Wih^T + bih -> buf[b*2048 + j], j in [0,1536) ---
        for (int nt = wv; nt < 96; nt += 32) {
            f8 acc = {};
            const _Float16* Wn = wih + (size_t)nt * 16 * 768;
#pragma unroll 4
            for (int kk = 0; kk < 768; kk += 32) {
                h16 a = load_a_frag(xin, 768, kk);
                h16 b = load_b_frag(Wn, 768, kk);
                acc = wmma_f16f32(a, b, acc);
            }
            int coln = lane & 15, gid = lane >> 4;
            int j = nt * 16 + coln;
            float bs = bih[j];
#pragma unroll
            for (int r = 0; r < 8; ++r) buf[(r + gid * 8) * 2048 + j] = acc[r] + bs;
        }
        __syncthreads();

        // --- E2: gh = h @ Whh^T + bhh; r/z gates summed, n-gate kept separate ---
        for (int nt = wv; nt < 96; nt += 32) {
            f8 acc = {};
            const _Float16* Wn = whh + (size_t)nt * 16 * HD;
#pragma unroll 4
            for (int kk = 0; kk < HD; kk += 32) {
                h16 a = load_a_frag(hf, HD, kk);
                h16 b = load_b_frag(Wn, HD, kk);
                acc = wmma_f16f32(a, b, acc);
            }
            int coln = lane & 15, gid = lane >> 4;
            int j = nt * 16 + coln;
            float bs = bhh[j];
#pragma unroll
            for (int r = 0; r < 8; ++r) {
                int bb = r + gid * 8;
                float v = acc[r] + bs;
                if (j < 1024) buf[bb * 2048 + j] += v;        // r,z: gi+gh
                else          buf[bb * 2048 + 512 + j] = v;   // ghn at [1536,2048)
            }
        }
        __syncthreads();

        // --- F: GRU update ---
        for (int idx = tid; idx < NB * HD; idx += 1024) {
            int bb = idx / HD, i = idx % HD;
            float* rb = buf + (size_t)bb * 2048;
            float r = 1.f / (1.f + expf(-rb[i]));
            float z = 1.f / (1.f + expf(-rb[512 + i]));
            float n = tanhf(rb[1024 + i] + r * rb[1536 + i]);
            float hn = (1.f - z) * n + z * h[idx];
            h[idx] = hn; hf[idx] = (_Float16)hn;
        }
        __syncthreads();

        // --- G: pred = h_new @ out_w^T + out_b ---
        for (int idx = tid; idx < NB * VOC; idx += 1024) {
            int bb = idx / VOC, v = idx % VOC;
            const float* wr = out_w + (size_t)v * HD;
            const float* hr = h + (size_t)bb * HD;
            float acc = out_b[v];
            for (int k = 0; k < HD; ++k) acc += hr[k] * wr[k];
            preds[((size_t)bb * (LSEQ - 1) + s) * VOC + v] = acc;
        }
        __syncthreads();
    }
}

// ================= host launcher =================
extern "C" void kernel_launch(void* const* d_in, const int* in_sizes, int n_in,
                              void* d_out, int out_size, void* d_ws, size_t ws_size,
                              hipStream_t stream) {
    (void)in_sizes; (void)n_in; (void)out_size; (void)ws_size;
    const float* spec    = (const float*)d_in[0];
    const int*   ttext   = (const int*)d_in[1];
    const float* c1w = (const float*)d_in[2];  const float* c1b = (const float*)d_in[3];
    const float* g1  = (const float*)d_in[4];  const float* b1  = (const float*)d_in[5];
    const float* m1  = (const float*)d_in[6];  const float* v1  = (const float*)d_in[7];
    const float* c2w = (const float*)d_in[8];  const float* c2b = (const float*)d_in[9];
    const float* g2  = (const float*)d_in[10]; const float* b2  = (const float*)d_in[11];
    const float* m2  = (const float*)d_in[12]; const float* v2  = (const float*)d_in[13];
    const float* c3w = (const float*)d_in[14]; const float* c3b = (const float*)d_in[15];
    const float* g3  = (const float*)d_in[16]; const float* b3  = (const float*)d_in[17];
    const float* m3  = (const float*)d_in[18]; const float* v3  = (const float*)d_in[19];
    const float* efw = (const float*)d_in[20]; const float* efb = (const float*)d_in[21];
    const float* iw  = (const float*)d_in[22]; const float* ib  = (const float*)d_in[23];
    const float* aw  = (const float*)d_in[24]; const float* abv = (const float*)d_in[25];
    const float* vw  = (const float*)d_in[26]; const float* emb = (const float*)d_in[27];
    const float* wih = (const float*)d_in[28]; const float* whh = (const float*)d_in[29];
    const float* bih = (const float*)d_in[30]; const float* bhh = (const float*)d_in[31];
    const float* ow  = (const float*)d_in[32]; const float* ob  = (const float*)d_in[33];
    float* preds = (float*)d_out;

    // ---- workspace bump allocator (256B aligned) ----
    char* ws = (char*)d_ws;
    size_t off = 0;
    auto alloc = [&](size_t bytes) -> char* {
        char* p = ws + off; off += (bytes + 255) & ~(size_t)255; return p;
    };
    _Float16* w2h  = (_Float16*)alloc((size_t)64 * 288 * 2);
    _Float16* w3h  = (_Float16*)alloc((size_t)128 * 576 * 2);
    _Float16* efwh = (_Float16*)alloc((size_t)512 * 2048 * 2);
    _Float16* whT  = (_Float16*)alloc((size_t)512 * 512 * 2);   // attn Wh
    _Float16* weh  = (_Float16*)alloc((size_t)512 * 512 * 2);   // attn We
    _Float16* wihh = (_Float16*)alloc((size_t)1536 * 768 * 2);
    _Float16* whhh = (_Float16*)alloc((size_t)1536 * 512 * 2);
    _Float16* embh = (_Float16*)alloc((size_t)29 * 256 * 2);
    float*    em   = (float*)alloc((size_t)16 * 512 * 4);
    float*    h0b  = (float*)alloc((size_t)16 * 512 * 4);
    _Float16* encb = (_Float16*)alloc((size_t)3200 * 512 * 2);
    _Float16* act1 = (_Float16*)alloc((size_t)26214400 * 2);    // also conv3_out
    _Float16* c2o  = (_Float16*)alloc((size_t)52428800 * 2);    // also x (B,200,2048)
    _Float16* act2 = (_Float16*)alloc((size_t)13107200 * 2);    // also enc_proj f32
    _Float16* c3o  = act1;
    _Float16* xbuf = c2o;
    float*    epf  = (float*)act2;

    const int CT = 256;
    auto gsz = [](long n, int b) { return (unsigned)((n + b - 1) / b); };

    // ---- weight conversions ----
    cvt_kernel<<<gsz(64 * 288, CT), CT, 0, stream>>>(c2w, w2h, 64, 288, 288, 0);
    cvt_kernel<<<gsz(128 * 576, CT), CT, 0, stream>>>(c3w, w3h, 128, 576, 576, 0);
    cvt_kernel<<<gsz(512 * 2048, CT), CT, 0, stream>>>(efw, efwh, 512, 2048, 2048, 0);
    cvt_kernel<<<gsz(512 * 512, CT), CT, 0, stream>>>(aw, whT, 512, 512, 1024, 0);
    cvt_kernel<<<gsz(512 * 512, CT), CT, 0, stream>>>(aw, weh, 512, 512, 1024, 512);
    cvt_kernel<<<gsz(1536 * 768, CT), CT, 0, stream>>>(wih, wihh, 1536, 768, 768, 0);
    cvt_kernel<<<gsz(1536 * 512, CT), CT, 0, stream>>>(whh, whhh, 1536, 512, 512, 0);
    cvt_kernel<<<gsz(29 * 256, CT), CT, 0, stream>>>(emb, embh, 29, 256, 256, 0);

    // ---- encoder ----
    conv1_kernel<<<gsz((long)16 * 32 * 64 * 800, CT), CT, 0, stream>>>(
        spec, c1w, c1b, g1, b1, m1, v1, act1);

    // conv2: (16,32,64,800) -> (16,64,64,800), K=288, 4 waves/block, 9KB LDS tile
    conv_gemm_kernel<32, 64, 800><<<16 * 3200, 128, (size_t)16 * 288 * 2, stream>>>(
        act1, w2h, c2b, g2, b2, m2, v2, c2o, 64);
    pool_kernel<64, 64, 800><<<gsz((long)16 * 64 * 32 * 400, CT), CT, 0, stream>>>(c2o, act2);

    // conv3: (16,64,32,400) -> (16,128,32,400), K=576, 8 waves/block, 18KB LDS tile
    conv_gemm_kernel<64, 32, 400><<<16 * 800, 256, (size_t)16 * 576 * 2, stream>>>(
        act2, w3h, c3b, g3, b3, m3, v3, c3o, 128);
    pool3t_kernel<<<gsz((long)16 * 200 * 2048, CT), CT, 0, stream>>>(c3o, xbuf);

    // enc = x @ enc_fc_w^T + b : M=3200, N=512, K=2048 (64KB LDS A-tile via TDM)
    gemm16_kernel<512, 2048><<<200, 256, (size_t)16 * 2048 * 2, stream>>>(
        xbuf, efwh, efb, (float*)nullptr, encb);

    encmean_kernel<<<gsz(16 * 512, CT), CT, 0, stream>>>(encb, em);
    h0_kernel<<<gsz(16 * 512, CT), CT, 0, stream>>>(em, iw, ib, h0b);

    // enc_proj = enc @ We^T : M=3200, N=512, K=512 (16KB LDS A-tile via TDM)
    gemm16_kernel<512, 512><<<200, 256, (size_t)16 * 512 * 2, stream>>>(
        encb, weh, (const float*)nullptr, epf, (_Float16*)nullptr);

    // ---- persistent decoder (one WGP, 32 waves, ~213KB LDS) ----
    size_t smem = (size_t)NB * HD * 4       // h
                + (size_t)NB * HD * 2       // hf
                + (size_t)NB * TP * 4       // attw
                + (size_t)NB * 768 * 2      // xin
                + (size_t)NB * 2048 * 4;    // buf
    decoder_kernel<<<1, 1024, smem, stream>>>(
        encb, epf, h0b, whT, abv, vw, embh, ttext,
        wihh, whhh, bih, bhh, ow, ob, preds);
}
